// NPMILossWithDiversity_19387482375081
// MI455X (gfx1250) — matrix operand order repaired and checked by reading, not compiled, not run on hardware
//
#include <hip/hip_runtime.h>
#include <hip/hip_bf16.h>

#define VOCAB 10000
#define NT    200
#define NTP   208        // padded to 13*16 topic rows
#define TOPK  20
#define GROUPS 13
#define GROWS  320       // 16 topics * 20 gathered rows per group
#define KTILES 10        // 320 / 32

typedef __attribute__((ext_vector_type(16))) __bf16 v16bf;
typedef __attribute__((ext_vector_type(8)))  float  v8f;

// ---------------- init: zero col counts + output scalar ----------------
__global__ void k_init(int* __restrict__ colCnt, float* __restrict__ out) {
  int i = blockIdx.x * 256 + threadIdx.x;
  if (i < VOCAB) colCnt[i] = 0;
  if (i == 0) out[0] = 0.0f;
}

// ---------------- transpose jacobian [V,200] -> betaT [208,V] ----------------
__global__ void k_transpose(const float* __restrict__ jac, float* __restrict__ betaT) {
  __shared__ float tile[32][33];
  int v0 = blockIdx.x * 32;
  int t0 = blockIdx.y * 32;
  int tx = threadIdx.x, ty = threadIdx.y;           // 32x8
  for (int i = ty; i < 32; i += 8) {
    int v = v0 + i, t = t0 + tx;
    tile[i][tx] = (v < VOCAB && t < NT) ? jac[(size_t)v * NT + t] : 0.0f;
  }
  __syncthreads();
  for (int i = ty; i < 32; i += 8) {
    int t = t0 + i, v = v0 + tx;
    if (t < NTP && v < VOCAB) betaT[(size_t)t * VOCAB + v] = tile[tx][i];
  }
}

// ---------------- per-topic: row softmax stats + iterative top-20 ----------------
__global__ void k_topk(const float* __restrict__ betaT, int* __restrict__ topkIdx,
                       float* __restrict__ topkW, float* __restrict__ rowMax,
                       float* __restrict__ rowSum, int* __restrict__ colCnt) {
  int t = blockIdx.x;
  __shared__ float s_row[VOCAB];                    // 40KB of 320KB WGP LDS
  __shared__ float s_rv[256];
  __shared__ int   s_ri[256];
  __shared__ float s_tv[TOPK];
  __shared__ int   s_ti[TOPK];
  int tid = threadIdx.x;
  const float* row = betaT + (size_t)t * VOCAB;
  for (int v = tid; v < VOCAB; v += 256) s_row[v] = row[v];
  __syncthreads();
  // row max
  float m = -3.0e38f;
  for (int v = tid; v < VOCAB; v += 256) m = fmaxf(m, s_row[v]);
  s_rv[tid] = m; __syncthreads();
  for (int s = 128; s > 0; s >>= 1) {
    if (tid < s) s_rv[tid] = fmaxf(s_rv[tid], s_rv[tid + s]);
    __syncthreads();
  }
  float rmx = s_rv[0]; __syncthreads();
  // softmax denominator over the full row
  float se = 0.0f;
  for (int v = tid; v < VOCAB; v += 256) se += __expf(s_row[v] - rmx);
  s_rv[tid] = se; __syncthreads();
  for (int s = 128; s > 0; s >>= 1) {
    if (tid < s) s_rv[tid] += s_rv[tid + s];
    __syncthreads();
  }
  if (tid == 0) { rowMax[t] = rmx; rowSum[t] = s_rv[0]; }
  __syncthreads();
  // iterative top-k: 20 block-wide argmax passes with masking
  for (int it = 0; it < TOPK; ++it) {
    float bv = -3.0e38f; int bi = 0;
    for (int v = tid; v < VOCAB; v += 256) {
      float x = s_row[v];
      if (x > bv) { bv = x; bi = v; }
    }
    s_rv[tid] = bv; s_ri[tid] = bi; __syncthreads();
    for (int s = 128; s > 0; s >>= 1) {
      if (tid < s && s_rv[tid + s] > s_rv[tid]) { s_rv[tid] = s_rv[tid + s]; s_ri[tid] = s_ri[tid + s]; }
      __syncthreads();
    }
    if (tid == 0) { s_tv[it] = s_rv[0]; s_ti[it] = s_ri[0]; s_row[s_ri[0]] = -3.0e38f; }
    __syncthreads();
  }
  // softmax over the 20 values (descending -> max is s_tv[0])
  if (tid == 0) {
    float sm = 0.0f;
    for (int k = 0; k < TOPK; ++k) sm += __expf(s_tv[k] - s_tv[0]);
    s_rv[0] = sm;
  }
  __syncthreads();
  if (tid < TOPK) {
    float w = (t < NT) ? (__expf(s_tv[tid] - s_tv[0]) / s_rv[0]) : 0.0f;
    topkIdx[t * TOPK + tid] = s_ti[tid];
    topkW[t * TOPK + tid]   = w;
    if (t < NT) atomicAdd(&colCnt[s_ti[tid]], 1);
  }
}

// ---------------- stacked-gather WMMA GEMM: scores[208,V] ----------------
// Per topic-group of 16: stack the 320 gathered npmi rows as K; A is the
// (near block-diagonal) weight matrix. 10 K-tiles of V_WMMA_F32_16X16X32_BF16
// produce an exact 16x16 scores tile while reading only the gathered rows.
// Diagonal fix (npmi.diag = 1) is applied afterwards by k_diagfix, so the
// inner loop is a pure unconditional gather.
__global__ void k_scores(const float* __restrict__ npmi, const int* __restrict__ topkIdx,
                         const float* __restrict__ topkW, float* __restrict__ scores) {
  __shared__ __align__(16) int    s_gidx[GROWS];
  __shared__ float                s_gw[GROWS];
  __shared__ __align__(32) __bf16 s_afrag[KTILES * 32 * 16];   // fragment-ordered A (10KB)
  int g   = blockIdx.y;
  int tid = threadIdx.x;                            // 128 threads = 4 waves
  // stage 1: gather indices + weights for this topic group
  for (int i = tid; i < GROWS; i += 128) {
    int topic = g * 16 + i / TOPK;
    s_gidx[i] = topkIdx[topic * TOPK + i % TOPK];
    s_gw[i]   = topkW [topic * TOPK + i % TOPK];
  }
  __syncthreads();
  // stage 2: build A fragments in fragment order: element ((k*32+lane)*16 + j)
  for (int i = tid; i < KTILES * 32 * 16; i += 128) {
    int j    = i & 15;
    int lane = (i >> 4) & 31;
    int k    = i >> 9;
    int half = lane >> 4, lmod = lane & 15;
    int kk = j + (half ? 16 : 0);  // contiguous K order (16*half + j), same order as the B gather
    int ga = k * 32 + kk;
    float av = ((ga / TOPK) == lmod) ? s_gw[ga] : 0.0f;
    s_afrag[i] = (__bf16)av;
  }
  __syncthreads();
  int wave = tid >> 5;
  int lane = tid & 31;
  int tile = blockIdx.x * 4 + wave;                 // 16-column N tile
  if (tile >= VOCAB / 16) return;                   // whole wave exits together
  int colBase = tile * 16;
  int half = lane >> 4;
  int lmod = lane & 15;
  int col  = colBase + lmod;                        // N index for B and C
  const float* nbase = npmi + col;
  v8f acc = {};
#pragma unroll
  for (int k = 0; k < KTILES; ++k) {
    // prefetch next K-tile's gathered rows (global_prefetch_b8)
    if (k + 1 < KTILES) {
      int pr = s_gidx[(k + 1) * 32 + lane];
      __builtin_prefetch(npmi + (size_t)pr * VOCAB + colBase, 0, 1);
    }
    // this lane's 16 gather indices are contiguous: vector LDS loads
    const int4* gp = (const int4*)&s_gidx[k * 32 + half * 16];
    int4 q0 = gp[0], q1 = gp[1], q2 = gp[2], q3 = gp[3];
    int idx[16] = { q0.x, q0.y, q0.z, q0.w, q1.x, q1.y, q1.z, q1.w,
                    q2.x, q2.y, q2.z, q2.w, q3.x, q3.y, q3.z, q3.w };
    // A fragment: two ds_load_b128
    v16bf a = *(const v16bf*)&s_afrag[(k * 32 + lane) * 16];
    // B fragment: 16 unconditional gathered loads (coalesced across lanes)
    float bf[16];
#pragma unroll
    for (int j = 0; j < 16; ++j) bf[j] = nbase[(size_t)idx[j] * VOCAB];
    v16bf b;
#pragma unroll
    for (int j = 0; j < 16; ++j) b[j] = (__bf16)bf[j];
    acc = __builtin_amdgcn_wmma_f32_16x16x32_bf16(false, a, false, b,
                                                  (short)0, acc, false, false);
  }
  // C 16x16 f32 layout: VGPR r -> M = r + half*8, N = lane%16
#pragma unroll
  for (int r = 0; r < 8; ++r) {
    int t = g * 16 + r + half * 8;
    scores[(size_t)t * VOCAB + col] = acc[r];
  }
}

// ---------------- diagonal correction: npmi.diag treated as 1.0 ----------------
// scores[t, idx] += w * (1 - npmi[idx, idx]); indices distinct per topic.
__global__ void k_diagfix(const float* __restrict__ npmi, const int* __restrict__ topkIdx,
                          const float* __restrict__ topkW, float* __restrict__ scores) {
  int i = blockIdx.x * 256 + threadIdx.x;           // over NT*TOPK
  if (i >= NT * TOPK) return;
  int t   = i / TOPK;
  int idx = topkIdx[i];
  float w = topkW[i];
  float d = npmi[(size_t)idx * VOCAB + idx];
  scores[(size_t)t * VOCAB + idx] += w * (1.0f - d);
}

// ---------------- per-row min/max of scores ----------------
__global__ void k_minmax(const float* __restrict__ scores, float* __restrict__ rMin,
                         float* __restrict__ rMax) {
  int t = blockIdx.x;
  __shared__ float smn[256], smx[256];
  int tid = threadIdx.x;
  float mn = 3.0e38f, mx = -3.0e38f;
  const float* row = scores + (size_t)t * VOCAB;
  for (int v = tid; v < VOCAB; v += 256) {
    float x = row[v];
    mn = fminf(mn, x); mx = fmaxf(mx, x);
  }
  smn[tid] = mn; smx[tid] = mx; __syncthreads();
  for (int s = 128; s > 0; s >>= 1) {
    if (tid < s) { smn[tid] = fminf(smn[tid], smn[tid + s]); smx[tid] = fmaxf(smx[tid], smx[tid + s]); }
    __syncthreads();
  }
  if (tid == 0) { rMin[t] = smn[0]; rMax[t] = smx[0]; }
}

// ---------------- fused weighted reduction -> scalar loss ----------------
__global__ void k_loss(const float* __restrict__ betaT, const float* __restrict__ scores,
                       const float* __restrict__ rMin, const float* __restrict__ rMax,
                       const float* __restrict__ rowMax, const float* __restrict__ rowSum,
                       const int* __restrict__ topkIdx, const int* __restrict__ colCnt,
                       float* __restrict__ out) {
  int t = blockIdx.x;                               // 0..199
  __shared__ int   s_ti[TOPK];
  __shared__ float s_red[256];
  int tid = threadIdx.x;
  if (tid < TOPK) s_ti[tid] = topkIdx[t * TOPK + tid];
  __syncthreads();
  float rmn = rMin[t], rmx = rMax[t];
  float inv  = 1.0f / (rmx - rmn);
  float bmax = rowMax[t];
  float binv = 1.0f / rowSum[t];
  float acc = 0.0f;
  for (int v = tid; v < VOCAB; v += 256) {
    float wn = 1.0f - (scores[(size_t)t * VOCAB + v] - rmn) * inv;
    float sb = __expf(betaT[(size_t)t * VOCAB + v] - bmax) * binv;
    float c  = 100.0f * sb * sb * wn;               // NPMI_SCALE
    int m = 0;
#pragma unroll
    for (int k = 0; k < TOPK; ++k) m |= (s_ti[k] == v);
    int div = (colCnt[v] - m) > 0;                  // selected by any OTHER topic
    acc += (div ? 0.7f : 0.3f) * c;                 // NPMI_LAMBDA split
  }
  s_red[tid] = acc; __syncthreads();
  for (int s = 128; s > 0; s >>= 1) {
    if (tid < s) s_red[tid] += s_red[tid + s];
    __syncthreads();
  }
  if (tid == 0) atomicAdd(out, 2.0f * s_red[0]);    // loss * 2
}

extern "C" void kernel_launch(void* const* d_in, const int* in_sizes, int n_in,
                              void* d_out, int out_size, void* d_ws, size_t ws_size,
                              hipStream_t stream) {
  const float* jac  = (const float*)d_in[0];        // [10000, 200]
  const float* npmi = (const float*)d_in[1];        // [10000, 10000]
  float* out = (float*)d_out;

  char* ws = (char*)d_ws;
  size_t off = 0;
  auto alloc = [&](size_t bytes) -> void* {
    void* p = ws + off;
    off = (off + bytes + 255) & ~(size_t)255;
    return p;
  };
  float* betaT   = (float*)alloc((size_t)NTP * VOCAB * sizeof(float));
  float* scores  = (float*)alloc((size_t)NTP * VOCAB * sizeof(float));
  int*   topkIdx = (int*)  alloc((size_t)NTP * TOPK * sizeof(int));
  float* topkW   = (float*)alloc((size_t)NTP * TOPK * sizeof(float));
  float* rowMax  = (float*)alloc(NTP * sizeof(float));
  float* rowSum  = (float*)alloc(NTP * sizeof(float));
  float* rMin    = (float*)alloc(NTP * sizeof(float));
  float* rMax    = (float*)alloc(NTP * sizeof(float));
  int*   colCnt  = (int*)  alloc(VOCAB * sizeof(int));
  (void)ws_size; (void)in_sizes; (void)n_in; (void)out_size;

  k_init<<<(VOCAB + 255) / 256, 256, 0, stream>>>(colCnt, out);
  k_transpose<<<dim3((VOCAB + 31) / 32, (NTP + 31) / 32), dim3(32, 8), 0, stream>>>(jac, betaT);
  k_topk<<<NTP, 256, 0, stream>>>(betaT, topkIdx, topkW, rowMax, rowSum, colCnt);
  k_scores<<<dim3((VOCAB / 16 + 3) / 4, GROUPS), 128, 0, stream>>>(npmi, topkIdx, topkW, scores);
  k_diagfix<<<(NT * TOPK + 255) / 256, 256, 0, stream>>>(npmi, topkIdx, topkW, scores);
  k_minmax<<<NT, 256, 0, stream>>>(scores, rMin, rMax);
  k_loss<<<NT, 256, 0, stream>>>(betaT, scores, rMin, rMax, rowMax, rowSum, topkIdx, colCnt, out);
}
